// Attention_40827959116229
// MI455X (gfx1250) — matrix operand order repaired and checked by reading, not compiled, not run on hardware
//
#include <hip/hip_runtime.h>

// ---------------------------------------------------------------------------
// MI455X (gfx1250) multi-head attention, bf16 WMMA pipeline, round 2.
//   Stage 0: one-time f32 -> bf16 conversion of activations + weights
//            (memory-bound, ~3us; removes all cvt VALU from GEMM loops)
//   Stage 1: q/k/v projections (bf16 WMMA, b128 fragment loads;
//            V written transposed [b,h,hd,s] for the PV matmul)
//   Stage 2: flash attention (WMMA scores + online softmax + WMMA PV)
//   Stage 3: output projection (bf16 WMMA, f32 out)
// ---------------------------------------------------------------------------

typedef __attribute__((ext_vector_type(16))) __bf16        v16bf;
typedef __attribute__((ext_vector_type(8)))  float         v8f;
typedef __attribute__((ext_vector_type(8)))  unsigned int  v8u;
typedef __attribute__((ext_vector_type(4)))  unsigned int  v4u;
typedef __attribute__((ext_vector_type(4)))  float         v4f;

namespace {
constexpr int Bn  = 4;
constexpr int Sn  = 1024;
constexpr int Dn  = 1024;
constexpr int Hn  = 16;
constexpr int HDn = 64;
constexpr int Mn  = Bn * Sn;            // 4096 rows
constexpr float SCALE = 0.125f;         // 1/sqrt(64)
}

// f32 -> bf16 via native conversion (hardware cvt on gfx1250)
static __device__ __forceinline__ unsigned short bf16_bits(float x) {
  __bf16 b = (__bf16)x;
  return __builtin_bit_cast(unsigned short, b);
}
static __device__ __forceinline__ unsigned int cvt_pk(float lo, float hi) {
  return (unsigned int)bf16_bits(lo) | ((unsigned int)bf16_bits(hi) << 16);
}
static __device__ __forceinline__ v8f wmma_bf16(v8u a, v8u b, v8f c) {
  return __builtin_amdgcn_wmma_f32_16x16x32_bf16(
      false, __builtin_bit_cast(v16bf, a),
      false, __builtin_bit_cast(v16bf, b),
      (short)0, c, false, false);
}
// Fragment = two contiguous 16B runs (per ISA 7.12.2 bf16 A/B layouts)
static __device__ __forceinline__ v8u load_frag(const unsigned short* p0,
                                                const unsigned short* p1) {
  v4u lo = *(const v4u*)p0;
  v4u hi = *(const v4u*)p1;
  v8u r;
  r[0] = lo[0]; r[1] = lo[1]; r[2] = lo[2]; r[3] = lo[3];
  r[4] = hi[0]; r[5] = hi[1]; r[6] = hi[2]; r[7] = hi[3];
  return r;
}

// ---------------------------------------------------------------------------
// Stage 0: elementwise f32 -> packed bf16 (8 elements / thread)
// ---------------------------------------------------------------------------
__global__ __launch_bounds__(256)
void cvt_bf16_kernel(const float* __restrict__ src,
                     unsigned short* __restrict__ dst, int n) {
  const int i = (blockIdx.x * 256 + threadIdx.x) * 8;
  if (i >= n) return;
  v4f f0 = *(const v4f*)(src + i);
  v4f f1 = *(const v4f*)(src + i + 4);
  v4u o;
  o[0] = cvt_pk(f0[0], f0[1]);
  o[1] = cvt_pk(f0[2], f0[3]);
  o[2] = cvt_pk(f1[0], f1[1]);
  o[3] = cvt_pk(f1[2], f1[3]);
  *(v4u*)(dst + i) = o;
}

// ---------------------------------------------------------------------------
// Stage 1: projection GEMM, all-bf16 operands.
//   Y[m,n] = (sum_k X[m,k]*W[n,k] + bias[n]) * scale
// 64x64 tile per wave; 8 A-b128 + 8 B-b128 loads per 16 WMMAs.
// transposeOut: 0 -> Y row-major [m, n] (stride N)
//               1 -> Y = vt[(b*Dn + n)*Sn + s]  (s-contiguous, b128 stores)
// grid = (N/64, M/256), block = 128.
// ---------------------------------------------------------------------------
__global__ __launch_bounds__(128)
void proj_kernel(const unsigned short* __restrict__ X,
                 const unsigned short* __restrict__ W,
                 const float* __restrict__ bias,
                 unsigned short* __restrict__ Y,
                 int M, int N, int K, float scale, int transposeOut) {
  const int lane = threadIdx.x & 31;
  const int wave = threadIdx.x >> 5;
  const int l16  = lane & 15;
  const int half = lane >> 4;
  const int m0   = blockIdx.y * 256 + wave * 64;
  const int n0   = blockIdx.x * 64;

  v8f acc[4][4];
#pragma unroll
  for (int i = 0; i < 4; ++i)
#pragma unroll
    for (int j = 0; j < 4; ++j) acc[i][j] = (v8f)0.0f;

  const unsigned short* xr0 = X + (size_t)(m0 + l16) * K;
  const unsigned short* wr0 = W + (size_t)(n0 + l16) * K;

  for (int k0 = 0; k0 < K; k0 += 32) {
    v8u a[4], b[4];
#pragma unroll
    for (int i = 0; i < 4; ++i) {
      const unsigned short* xr = xr0 + (size_t)i * 16 * K + k0;
      a[i] = load_frag(xr + half * 8, xr + 16 + half * 8);
    }
#pragma unroll
    for (int j = 0; j < 4; ++j) {
      const unsigned short* wr = wr0 + (size_t)j * 16 * K + k0;
      b[j] = load_frag(wr + half * 8, wr + 16 + half * 8);
    }
#pragma unroll
    for (int i = 0; i < 4; ++i)
#pragma unroll
      for (int j = 0; j < 4; ++j)
        acc[i][j] = wmma_bf16(a[i], b[j], acc[i][j]);
  }

  if (!transposeOut) {
#pragma unroll
    for (int j = 0; j < 4; ++j) {
      const int n   = n0 + j * 16 + l16;
      const float bb = bias[n];
#pragma unroll
      for (int i = 0; i < 4; ++i)
#pragma unroll
        for (int v = 0; v < 8; ++v) {
          const int m = m0 + i * 16 + half * 8 + v;
          Y[(size_t)m * N + n] = bf16_bits((acc[i][j][v] + bb) * scale);
        }
    }
  } else {
    // vt layout: row (b*Dn + n), column s; 8 consecutive s per lane -> b128
    const int bidx = m0 / Sn;            // whole 256-row block is in one b
#pragma unroll
    for (int j = 0; j < 4; ++j) {
      const int n   = n0 + j * 16 + l16;
      const float bb = bias[n];
      unsigned short* yr = Y + ((size_t)bidx * Dn + n) * Sn;
#pragma unroll
      for (int i = 0; i < 4; ++i) {
        const int s0 = m0 + i * 16 + half * 8 - bidx * Sn;
        v4u o;
#pragma unroll
        for (int p = 0; p < 4; ++p)
          o[p] = cvt_pk((acc[i][j][2 * p] + bb) * scale,
                        (acc[i][j][2 * p + 1] + bb) * scale);
        *(v4u*)(yr + s0) = o;
      }
    }
  }
}

// ---------------------------------------------------------------------------
// Stage 3: output GEMM, bf16 A (ctx) x bf16 W -> f32 out + bias
// ---------------------------------------------------------------------------
__global__ __launch_bounds__(128)
void oproj_kernel(const unsigned short* __restrict__ A,
                  const unsigned short* __restrict__ W,
                  const float* __restrict__ bias,
                  float* __restrict__ out, int M, int N, int K) {
  const int lane = threadIdx.x & 31;
  const int wave = threadIdx.x >> 5;
  const int l16  = lane & 15;
  const int half = lane >> 4;
  const int m0   = blockIdx.y * 256 + wave * 64;
  const int n0   = blockIdx.x * 64;

  v8f acc[4][4];
#pragma unroll
  for (int i = 0; i < 4; ++i)
#pragma unroll
    for (int j = 0; j < 4; ++j) acc[i][j] = (v8f)0.0f;

  const unsigned short* ar0 = A + (size_t)(m0 + l16) * K;
  const unsigned short* wr0 = W + (size_t)(n0 + l16) * K;

  for (int k0 = 0; k0 < K; k0 += 32) {
    v8u a[4], b[4];
#pragma unroll
    for (int i = 0; i < 4; ++i) {
      const unsigned short* ar = ar0 + (size_t)i * 16 * K + k0;
      a[i] = load_frag(ar + half * 8, ar + 16 + half * 8);
    }
#pragma unroll
    for (int j = 0; j < 4; ++j) {
      const unsigned short* wr = wr0 + (size_t)j * 16 * K + k0;
      b[j] = load_frag(wr + half * 8, wr + 16 + half * 8);
    }
#pragma unroll
    for (int i = 0; i < 4; ++i)
#pragma unroll
      for (int j = 0; j < 4; ++j)
        acc[i][j] = wmma_bf16(a[i], b[j], acc[i][j]);
  }

#pragma unroll
  for (int j = 0; j < 4; ++j) {
    const int n   = n0 + j * 16 + l16;
    const float bb = bias[n];
#pragma unroll
    for (int i = 0; i < 4; ++i)
#pragma unroll
      for (int v = 0; v < 8; ++v) {
        const int m = m0 + i * 16 + half * 8 + v;
        out[(size_t)m * N + n] = acc[i][j][v] + bb;
      }
  }
}

// ---------------------------------------------------------------------------
// Stage 2: flash attention. One (b,h) + 64 query rows per block, 16 per wave.
// qh/kh row-major [B,S,D] bf16 (head slice = 64 contiguous cols);
// vt transposed [(b*Dn + h*64+hd) * Sn + s] bf16. ctx row-major bf16.
// grid = (B*H, S/64), block = 128.
// ---------------------------------------------------------------------------
__global__ __launch_bounds__(128)
void attn_kernel(const unsigned short* __restrict__ Qh,
                 const unsigned short* __restrict__ Kh,
                 const unsigned short* __restrict__ Vt,
                 unsigned short* __restrict__ Ctx) {
  __shared__ unsigned short p_lds[4][16][32];  // per-wave 16x32 bf16 P tile

  const int lane = threadIdx.x & 31;
  const int wave = threadIdx.x >> 5;
  const int l16  = lane & 15;
  const int half = lane >> 4;
  const int bh   = blockIdx.x;       // 0..63
  const int b    = bh >> 4;
  const int h    = bh & 15;
  const int q0   = blockIdx.y * 64 + wave * 16;
  const int col0 = h * HDn;

  // Q fragments: A-layout, two K=32 steps over HD=64 (2x b128 each)
  v8u aq[2];
  {
    const unsigned short* qr = Qh + (size_t)((b * Sn + q0 + l16) * Dn + col0);
#pragma unroll
    for (int st = 0; st < 2; ++st)
      aq[st] = load_frag(qr + st * 32 + half * 8, qr + st * 32 + 16 + half * 8);
  }

  float rmax[8], rsum[8];
#pragma unroll
  for (int v = 0; v < 8; ++v) { rmax[v] = -3.0e38f; rsum[v] = 0.0f; }
  v8f acc[4];
#pragma unroll
  for (int t = 0; t < 4; ++t) acc[t] = (v8f)0.0f;

  for (int jp = 0; jp < Sn / 32; ++jp) {      // 32 keys per outer step
#pragma unroll
    for (int jj = 0; jj < 2; ++jj) {          // two 16-key score tiles
      const int kt0 = jp * 32 + jj * 16;
      // K fragment: B-layout (lane = key column, k-dim = hd, contiguous)
      const unsigned short* kr =
          Kh + (size_t)((b * Sn + kt0 + l16) * Dn + col0);
      v8u bk0 = load_frag(kr + half * 8, kr + 16 + half * 8);
      v8u bk1 = load_frag(kr + 32 + half * 8, kr + 32 + 16 + half * 8);

      v8f sc = (v8f)0.0f;
      sc = wmma_bf16(aq[0], bk0, sc);
      sc = wmma_bf16(aq[1], bk1, sc);

      // online softmax: lane = key col, VGPR v -> q-row half*8+v
#pragma unroll
      for (int v = 0; v < 8; ++v) {
        const float s = sc[v];
        float tm = s;
#pragma unroll
        for (int off = 8; off > 0; off >>= 1)
          tm = fmaxf(tm, __shfl_xor(tm, off, 16));
        const float nm = fmaxf(rmax[v], tm);
        const float f  = __expf(rmax[v] - nm);
        rmax[v] = nm;
        const float p = __expf(s - nm);
        float ps = p;
#pragma unroll
        for (int off = 8; off > 0; off >>= 1)
          ps += __shfl_xor(ps, off, 16);
        rsum[v] = rsum[v] * f + ps;
#pragma unroll
        for (int t = 0; t < 4; ++t) acc[t][v] *= f;
        p_lds[wave][half * 8 + v][jj * 16 + l16] = bf16_bits(p);
      }
    }

    // wave-private LDS round-trip: C-layout -> A-layout for P (16x32 keys)
    asm volatile("s_wait_dscnt 0" ::: "memory");
    v8u ap = load_frag(&p_lds[wave][l16][half * 8],
                       &p_lds[wave][l16][16 + half * 8]);

    // V fragments from transposed layout: lane = hd column, k-dim = key
    // (keys half*8..+7 and 16+half*8..+7 per lane -> 2x b128)
#pragma unroll
    for (int t = 0; t < 4; ++t) {
      const unsigned short* vr =
          Vt + ((size_t)b * Dn + col0 + t * 16 + l16) * Sn + jp * 32;
      v8u bv = load_frag(vr + half * 8, vr + 16 + half * 8);
      acc[t] = wmma_bf16(ap, bv, acc[t]);
    }
  }

  // normalize, store ctx (bf16, [B,S,D] head-merged layout)
#pragma unroll
  for (int v = 0; v < 8; ++v) {
    const float inv = 1.0f / rsum[v];
    const int m = q0 + half * 8 + v;
#pragma unroll
    for (int t = 0; t < 4; ++t) {
      const int n = col0 + t * 16 + l16;
      Ctx[(size_t)(b * Sn + m) * Dn + n] = bf16_bits(acc[t][v] * inv);
    }
  }
}

// ---------------------------------------------------------------------------
extern "C" void kernel_launch(void* const* d_in, const int* in_sizes, int n_in,
                              void* d_out, int out_size, void* d_ws,
                              size_t ws_size, hipStream_t stream) {
  const float* q  = (const float*)d_in[0];
  const float* k  = (const float*)d_in[1];
  const float* v  = (const float*)d_in[2];
  const float* Wq = (const float*)d_in[3];
  const float* bq = (const float*)d_in[4];
  const float* Wk = (const float*)d_in[5];
  const float* bk = (const float*)d_in[6];
  const float* Wv = (const float*)d_in[7];
  const float* bv = (const float*)d_in[8];
  const float* Wo = (const float*)d_in[9];
  const float* bo = (const float*)d_in[10];
  float* out = (float*)d_out;

  // workspace layout (ushort elements): 56 MB total
  const size_t ACT = (size_t)Mn * Dn;   // 4M elems
  const size_t WGT = (size_t)Dn * Dn;   // 1M elems
  unsigned short* ws  = (unsigned short*)d_ws;
  unsigned short* qb  = ws;                       // bf16 copies of inputs
  unsigned short* kb  = ws + ACT;
  unsigned short* vb  = ws + 2 * ACT;
  unsigned short* Wqb = ws + 3 * ACT;
  unsigned short* Wkb = Wqb + WGT;
  unsigned short* Wvb = Wqb + 2 * WGT;
  unsigned short* Wob = Wqb + 3 * WGT;
  unsigned short* qh  = Wqb + 4 * WGT;            // projected Q (scaled)
  unsigned short* kh  = qh + ACT;                 // projected K
  unsigned short* vt  = qh + 2 * ACT;             // projected V, transposed
  unsigned short* ctx = qb;  // reuse: qb dead once qh is built; attn reads its
                             // own Q rows before writing ctx anywhere

  const dim3 blk(128);

  // Stage 0: f32 -> bf16
  {
    const int na = (int)ACT, nw = (int)WGT;
    const dim3 ga((na / 8 + 255) / 256), gw((nw / 8 + 255) / 256);
    cvt_bf16_kernel<<<ga, dim3(256), 0, stream>>>(q, qb, na);
    cvt_bf16_kernel<<<ga, dim3(256), 0, stream>>>(k, kb, na);
    cvt_bf16_kernel<<<ga, dim3(256), 0, stream>>>(v, vb, na);
    cvt_bf16_kernel<<<gw, dim3(256), 0, stream>>>(Wq, Wqb, nw);
    cvt_bf16_kernel<<<gw, dim3(256), 0, stream>>>(Wk, Wkb, nw);
    cvt_bf16_kernel<<<gw, dim3(256), 0, stream>>>(Wv, Wvb, nw);
    cvt_bf16_kernel<<<gw, dim3(256), 0, stream>>>(Wo, Wob, nw);
  }

  // Stage 1: projections
  const dim3 gproj(Dn / 64, Mn / 256);   // (16, 16)
  proj_kernel<<<gproj, blk, 0, stream>>>(qb, Wqb, bq, qh, Mn, Dn, Dn, SCALE, 0);
  proj_kernel<<<gproj, blk, 0, stream>>>(kb, Wkb, bk, kh, Mn, Dn, Dn, 1.0f, 0);
  proj_kernel<<<gproj, blk, 0, stream>>>(vb, Wvb, bv, vt, Mn, Dn, Dn, 1.0f, 1);

  // Stage 2: attention
  const dim3 gattn(Bn * Hn, Sn / 64);    // (64, 16)
  attn_kernel<<<gattn, blk, 0, stream>>>(qh, kh, vt, ctx);

  // Stage 3: output projection
  oproj_kernel<<<gproj, blk, 0, stream>>>(ctx, Wob, bo, out, Mn, Dn, Dn);
}
